// RecurrentModule_56642028699764
// MI455X (gfx1250) — matrix operand order repaired and checked by reading, not compiled, not run on hardware
//
#include <hip/hip_runtime.h>
#include <hip/hip_bf16.h>
#include <stdint.h>

typedef __bf16 bf16_t;
typedef __attribute__((ext_vector_type(16))) __bf16 v16bf;
typedef __attribute__((ext_vector_type(8)))  __bf16 v8bf;
typedef __attribute__((ext_vector_type(8)))  float  v8f;

#define B_   16
#define S_   1024
#define D_   512
#define H_   1024
#define G4_  4096
#define BS_  (B_*S_)

// ---------------- WMMA fragment helpers (wave32, 16x16x32 bf16) ----------------
// A (16xK tile, row-major memory): lane half 0 -> K 0..15 contiguous, half 1 -> K 16..31 contiguous
__device__ __forceinline__ v16bf load_a_frag(const bf16_t* p, int ld) {
  int lane = threadIdx.x & 31;
  int m = lane & 15, half = lane >> 4;
  const bf16_t* base = p + (size_t)m * ld + half * 16;
  v8bf lo = *(const v8bf*)base;         // K = half*16 .. half*16+7
  v8bf hi = *(const v8bf*)(base + 8);   // K = half*16+8 ..
  v16bf r;
#pragma unroll
  for (int i = 0; i < 8; ++i) { r[i] = lo[i]; r[i + 8] = hi[i]; }
  return r;
}

// B (KxN tile) sourced from W[N x K] row-major (i.e. B = W^T):
// lane half 0 -> K 0..15 contiguous, half 1 -> K 16..31 contiguous (one 32B load)
__device__ __forceinline__ v16bf load_b_frag(const bf16_t* p, int ld) {
  int lane = threadIdx.x & 31;
  int n = lane & 15, half = lane >> 4;
  return *(const v16bf*)(p + (size_t)n * ld + half * 16);
}

__device__ __forceinline__ v8f wmma_bf16(v16bf a, v16bf b, v8f c) {
  return __builtin_amdgcn_wmma_f32_16x16x32_bf16(false, a, false, b, (short)0, c, false, false);
}

__device__ __forceinline__ float sigmoidf_fast(float x) {
  return 1.0f / (1.0f + __expf(-x));
}
__device__ __forceinline__ float tanhf_fast(float x) {
  x = fminf(fmaxf(x, -15.0f), 15.0f);
  float e = __expf(2.0f * x);
  return (e - 1.0f) / (e + 1.0f);
}

// ---------------- Generic bf16 WMMA GEMM: C = scale*(A @ W^T) + bias ----------------
// A: [M x K] bf16 row-major (lda), W: [N x K] bf16 row-major (ldb), C: fp32 or bf16 (ldc)
// Block = 256 thr = 8 waves (4x2), block tile 128x128, wave tile 32x64.
__global__ __launch_bounds__(256) void wmma_gemm_nt(
    const bf16_t* __restrict__ A, const bf16_t* __restrict__ Bm,
    const float* __restrict__ bias, float* __restrict__ Cf, bf16_t* __restrict__ Cb,
    int K, int lda, int ldb, int ldc,
    long long sA, long long sB, long long sC, float scale)
{
  int w  = threadIdx.x >> 5;
  int m0 = blockIdx.x * 128 + (w & 3) * 32;
  int n0 = blockIdx.y * 128 + (w >> 2) * 64;
  const bf16_t* Ab = A  + (long long)blockIdx.z * sA;
  const bf16_t* Bb = Bm + (long long)blockIdx.z * sB;

  v8f acc[2][4];
#pragma unroll
  for (int i = 0; i < 2; ++i)
#pragma unroll
    for (int j = 0; j < 4; ++j) { v8f z = {}; acc[i][j] = z; }

  for (int k0 = 0; k0 < K; k0 += 32) {
    if (k0 + 32 < K) {
      __builtin_prefetch(Bb + (size_t)n0 * ldb + k0 + 32, 0, 1);
      __builtin_prefetch(Ab + (size_t)m0 * lda + k0 + 32, 0, 1);
    }
    v16bf a[2];
    a[0] = load_a_frag(Ab + (size_t)m0        * lda + k0, lda);
    a[1] = load_a_frag(Ab + (size_t)(m0 + 16) * lda + k0, lda);
    v16bf b[4];
#pragma unroll
    for (int nt = 0; nt < 4; ++nt)
      b[nt] = load_b_frag(Bb + (size_t)(n0 + nt * 16) * ldb + k0, ldb);
#pragma unroll
    for (int mt = 0; mt < 2; ++mt)
#pragma unroll
      for (int nt = 0; nt < 4; ++nt)
        acc[mt][nt] = wmma_bf16(a[mt], b[nt], acc[mt][nt]);
  }

  int lane = threadIdx.x & 31;
  int cn = lane & 15, rb = (lane >> 4) * 8;
#pragma unroll
  for (int mt = 0; mt < 2; ++mt)
#pragma unroll
    for (int nt = 0; nt < 4; ++nt) {
      int n = n0 + nt * 16 + cn;
      float bv = bias ? bias[n] : 0.0f;
#pragma unroll
      for (int r = 0; r < 8; ++r) {
        int m = m0 + mt * 16 + rb + r;
        float v = acc[mt][nt][r] * scale + bv;
        long long off = (long long)blockIdx.z * sC + (long long)m * ldc + n;
        if (Cf) Cf[off] = v;
        else    Cb[off] = (bf16_t)v;
      }
    }
}

// ---------------- LSTM recurrent step ----------------
// gates = gx[:,t,:] + h_in @ Whh^T  ->  i,f,g,o  ->  c,h update
// grid 32 blocks x 256 thr; block owns 32 hidden cols across all 4 gates.
// wave w: gate g = w>>1, column sub-tile cc = w&1 -> one 16x16 WMMA tile (M=16 batches).
// Whh tiles are streamed into wave-private LDS double buffers with
// GLOBAL_LOAD_ASYNC_TO_LDS_B128 and s_wait_asynccnt, overlapping fetch with WMMA.
__global__ __launch_bounds__(256) void lstm_step(
    const float* __restrict__ gx, const bf16_t* __restrict__ Whh,
    const bf16_t* __restrict__ h_in, bf16_t* __restrict__ h_out,
    float* __restrict__ c_state, bf16_t* __restrict__ seq_out, int t)
{
  __shared__ float lds[16][4][32];
  __shared__ __align__(64) bf16_t stage[8][2][16][32];  // per-wave double buffer (1KB each)
  int tid = threadIdx.x;
  int w = tid >> 5, lane = tid & 31;
  int g = w >> 1, cc = w & 1;
  int wg = blockIdx.x;
  int n0 = g * H_ + wg * 32 + cc * 16;

  unsigned int sbase =
      (unsigned int)(uintptr_t)(__attribute__((address_space(3))) char*)&stage[w][0][0][0];

  // issue per-lane async copies of one 16x32 bf16 B-tile (rows n0..n0+15, k kk..kk+31)
  auto issue_stage = [&](int buf, int kk) {
#pragma unroll
    for (int j = 0; j < 2; ++j) {
      int chunk = j * 32 + lane;           // 64 x 16B chunks
      int row = chunk >> 2;
      int kb  = chunk & 3;
      unsigned long long ga =
          (unsigned long long)(uintptr_t)(Whh + (size_t)(n0 + row) * H_ + kk + kb * 8);
      unsigned int la = sbase + (unsigned int)(buf * 1024 + row * 64 + kb * 16);
      asm volatile("global_load_async_to_lds_b128 %0, %1, off"
                   :: "v"(la), "v"(ga) : "memory");
    }
  };

  v8f acc = {};
  issue_stage(0, 0);
  for (int k0 = 0; k0 < H_; k0 += 32) {
    int buf = (k0 >> 5) & 1;
    if (k0 + 32 < H_) {
      issue_stage(buf ^ 1, k0 + 32);
      asm volatile("s_wait_asynccnt 0x2" ::: "memory");  // current buf's 2 copies done
    } else {
      asm volatile("s_wait_asynccnt 0x0" ::: "memory");
    }
    v16bf a = load_a_frag(h_in + k0, H_);
    v16bf b = *(const v16bf*)(&stage[w][buf][lane & 15][(lane >> 4) * 16]);
    acc = wmma_bf16(a, b, acc);
  }

  int cn = lane & 15, rb = (lane >> 4) * 8;
#pragma unroll
  for (int r = 0; r < 8; ++r) {
    int bi  = rb + r;            // batch index (M row)
    int col = n0 + cn;           // column within 4H
    float v = acc[r] + gx[((size_t)bi * S_ + t) * G4_ + col];
    lds[bi][g][cc * 16 + cn] = v;
  }
  __syncthreads();

#pragma unroll
  for (int e = tid; e < 512; e += 256) {
    int m = e >> 5, jl = e & 31;
    int j = wg * 32 + jl;
    float ig = sigmoidf_fast(lds[m][0][jl]);
    float fg = sigmoidf_fast(lds[m][1][jl]);
    float gg = tanhf_fast  (lds[m][2][jl]);
    float og = sigmoidf_fast(lds[m][3][jl]);
    int sidx = m * H_ + j;
    float c = fg * c_state[sidx] + ig * gg;
    c_state[sidx] = c;
    float h = og * tanhf_fast(c);
    h_out[sidx] = (bf16_t)h;
    seq_out[((size_t)m * S_ + t) * H_ + j] = (bf16_t)h;
  }
}

// ---------------- Row softmax over 1024 cols -> bf16 probs ----------------
__global__ __launch_bounds__(256) void softmax_rows(
    const float* __restrict__ Sc, bf16_t* __restrict__ P)
{
  __shared__ float red[256];
  int row = blockIdx.x;
  int tid = threadIdx.x;
  const float* sr = Sc + (size_t)row * 1024;
  float v[4];
  float mx = -3.0e38f;
#pragma unroll
  for (int i = 0; i < 4; ++i) { v[i] = sr[tid + i * 256]; mx = fmaxf(mx, v[i]); }
  red[tid] = mx; __syncthreads();
  for (int s = 128; s > 0; s >>= 1) {
    if (tid < s) red[tid] = fmaxf(red[tid], red[tid + s]);
    __syncthreads();
  }
  mx = red[0]; __syncthreads();
  float sum = 0.0f;
#pragma unroll
  for (int i = 0; i < 4; ++i) { v[i] = __expf(v[i] - mx); sum += v[i]; }
  red[tid] = sum; __syncthreads();
  for (int s = 128; s > 0; s >>= 1) {
    if (tid < s) red[tid] += red[tid + s];
    __syncthreads();
  }
  float inv = 1.0f / red[0];
#pragma unroll
  for (int i = 0; i < 4; ++i)
    P[(size_t)row * 1024 + tid + i * 256] = (bf16_t)(v[i] * inv);
}

// ---------------- Batched [S,H] -> [H,S] bf16 transpose via LDS tiles ----------------
__global__ __launch_bounds__(256) void transpose_bf16(
    const bf16_t* __restrict__ in, bf16_t* __restrict__ out)
{
  __shared__ bf16_t tile[32][33];
  int b = blockIdx.z;
  int h0 = blockIdx.x * 32, s0 = blockIdx.y * 32;
  const bf16_t* inb  = in  + (size_t)b * S_ * H_;
  bf16_t*       outb = out + (size_t)b * S_ * H_;
#pragma unroll
  for (int i = 0; i < 4; ++i) {
    int s = threadIdx.y + i * 8;
    tile[s][threadIdx.x] = inb[(size_t)(s0 + s) * H_ + h0 + threadIdx.x];
  }
  __syncthreads();
#pragma unroll
  for (int i = 0; i < 4; ++i) {
    int h = threadIdx.y + i * 8;
    outb[(size_t)(h0 + h) * S_ + s0 + threadIdx.x] = tile[threadIdx.x][h];
  }
}

// ---------------- residual + LayerNorm(512) ----------------
__global__ __launch_bounds__(256) void add_layernorm(
    const float* __restrict__ proj, const float* __restrict__ x,
    const float* __restrict__ gamma, const float* __restrict__ beta,
    float* __restrict__ out)
{
  __shared__ float rs[256], rq[256];
  int row = blockIdx.x, tid = threadIdx.x;
  size_t base = (size_t)row * D_;
  float r0 = proj[base + tid]       + x[base + tid];
  float r1 = proj[base + tid + 256] + x[base + tid + 256];
  rs[tid] = r0 + r1;
  rq[tid] = r0 * r0 + r1 * r1;
  __syncthreads();
  for (int s = 128; s > 0; s >>= 1) {
    if (tid < s) { rs[tid] += rs[tid + s]; rq[tid] += rq[tid + s]; }
    __syncthreads();
  }
  float mu  = rs[0] * (1.0f / D_);
  float var = rq[0] * (1.0f / D_) - mu * mu;
  float is  = rsqrtf(var + 1e-5f);
  out[base + tid]       = (r0 - mu) * is * gamma[tid]       + beta[tid];
  out[base + tid + 256] = (r1 - mu) * is * gamma[tid + 256] + beta[tid + 256];
}

// ---------------- utilities ----------------
__global__ __launch_bounds__(256) void f2bf_kernel(
    const float* __restrict__ in, bf16_t* __restrict__ out, int n)
{
  for (int i = blockIdx.x * 256 + threadIdx.x; i < n; i += gridDim.x * 256)
    out[i] = (bf16_t)in[i];
}

__global__ __launch_bounds__(256) void addvec_kernel(
    const float* __restrict__ a, const float* __restrict__ b,
    float* __restrict__ o, int n)
{
  int i = blockIdx.x * 256 + threadIdx.x;
  if (i < n) o[i] = a[i] + b[i];
}

__global__ __launch_bounds__(256) void zero_u32(uint32_t* __restrict__ p, int n)
{
  for (int i = blockIdx.x * 256 + threadIdx.x; i < n; i += gridDim.x * 256)
    p[i] = 0u;
}

// ---------------- host orchestration ----------------
extern "C" void kernel_launch(void* const* d_in, const int* in_sizes, int n_in,
                              void* d_out, int out_size, void* d_ws, size_t ws_size,
                              hipStream_t stream) {
  (void)in_sizes; (void)n_in; (void)out_size; (void)ws_size;
  const float* xf    = (const float*)d_in[0];
  const float* Wih0  = (const float*)d_in[1];
  const float* Whh0  = (const float*)d_in[2];
  const float* bih0  = (const float*)d_in[3];
  const float* bhh0  = (const float*)d_in[4];
  const float* Wih1  = (const float*)d_in[5];
  const float* Whh1  = (const float*)d_in[6];
  const float* bih1  = (const float*)d_in[7];
  const float* bhh1  = (const float*)d_in[8];
  const float* Wq    = (const float*)d_in[9];
  const float* bq    = (const float*)d_in[10];
  const float* Wk    = (const float*)d_in[11];
  const float* bk    = (const float*)d_in[12];
  const float* Wv    = (const float*)d_in[13];
  const float* bv    = (const float*)d_in[14];
  const float* Wproj = (const float*)d_in[15];
  const float* gam   = (const float*)d_in[16];
  const float* bet   = (const float*)d_in[17];
  float* outp = (float*)d_out;

  // workspace carve-up (256B aligned)
  char* base = (char*)d_ws;
  size_t off = 0;
  auto take = [&](size_t bytes) -> char* {
    char* r = base + off;
    off = (off + bytes + 255) & ~(size_t)255;
    return r;
  };
  bf16_t* x_bf   = (bf16_t*)take((size_t)BS_ * D_ * 2);
  bf16_t* Wih0b  = (bf16_t*)take((size_t)G4_ * D_ * 2);
  bf16_t* Whh0b  = (bf16_t*)take((size_t)G4_ * H_ * 2);
  bf16_t* Wih1b  = (bf16_t*)take((size_t)G4_ * H_ * 2);
  bf16_t* Whh1b  = (bf16_t*)take((size_t)G4_ * H_ * 2);
  bf16_t* Wqb    = (bf16_t*)take((size_t)H_ * H_ * 2);
  bf16_t* Wkb    = (bf16_t*)take((size_t)H_ * H_ * 2);
  bf16_t* Wvb    = (bf16_t*)take((size_t)H_ * H_ * 2);
  bf16_t* Wprojb = (bf16_t*)take((size_t)D_ * H_ * 2);
  float*  bc0    = (float*)take((size_t)G4_ * 4);
  float*  bc1    = (float*)take((size_t)G4_ * 4);
  bf16_t* h0seq  = (bf16_t*)take((size_t)BS_ * H_ * 2);
  bf16_t* h1seq  = (bf16_t*)take((size_t)BS_ * H_ * 2);
  bf16_t* hA     = (bf16_t*)take((size_t)B_ * H_ * 2);   // ping
  bf16_t* hB     = (bf16_t*)take((size_t)B_ * H_ * 2);   // pong (contiguous with hA)
  float*  cst    = (float*)take((size_t)B_ * H_ * 4);    // contiguous with hB
  bf16_t* qb     = (bf16_t*)take((size_t)BS_ * H_ * 2);
  bf16_t* kb     = (bf16_t*)take((size_t)BS_ * H_ * 2);
  bf16_t* vb     = (bf16_t*)take((size_t)BS_ * H_ * 2);
  float*  gx     = (float*)take((size_t)BS_ * G4_ * 4);  // 256MB, reused post-LSTM:
  float*  scores = (float*)gx;                                            // 64MB
  bf16_t* attn   = (bf16_t*)((char*)gx + (size_t)64 * 1024 * 1024);       // 32MB
  bf16_t* vT     = (bf16_t*)((char*)gx + (size_t)96 * 1024 * 1024);       // 32MB
  bf16_t* ctx    = (bf16_t*)((char*)gx + (size_t)128 * 1024 * 1024);      // 32MB
  float*  projo  = (float*)((char*)gx + (size_t)160 * 1024 * 1024);       // 32MB

  dim3 blk(256);
  // --- fp32 -> bf16 conversions ---
  f2bf_kernel<<<2048, blk, 0, stream>>>(xf,    x_bf,   BS_ * D_);
  f2bf_kernel<<<2048, blk, 0, stream>>>(Wih0,  Wih0b,  G4_ * D_);
  f2bf_kernel<<<2048, blk, 0, stream>>>(Whh0,  Whh0b,  G4_ * H_);
  f2bf_kernel<<<2048, blk, 0, stream>>>(Wih1,  Wih1b,  G4_ * H_);
  f2bf_kernel<<<2048, blk, 0, stream>>>(Whh1,  Whh1b,  G4_ * H_);
  f2bf_kernel<<<2048, blk, 0, stream>>>(Wq,    Wqb,    H_ * H_);
  f2bf_kernel<<<2048, blk, 0, stream>>>(Wk,    Wkb,    H_ * H_);
  f2bf_kernel<<<2048, blk, 0, stream>>>(Wv,    Wvb,    H_ * H_);
  f2bf_kernel<<<1024, blk, 0, stream>>>(Wproj, Wprojb, D_ * H_);
  addvec_kernel<<<16, blk, 0, stream>>>(bih0, bhh0, bc0, G4_);
  addvec_kernel<<<16, blk, 0, stream>>>(bih1, bhh1, bc1, G4_);

  // --- LSTM layer 0 ---
  // gx = x @ Wih0^T + (bih0+bhh0)  : M=16384, N=4096, K=512
  wmma_gemm_nt<<<dim3(BS_ / 128, G4_ / 128, 1), blk, 0, stream>>>(
      x_bf, Wih0b, bc0, gx, nullptr, D_, D_, D_, G4_, 0, 0, 0, 1.0f);
  zero_u32<<<128, blk, 0, stream>>>((uint32_t*)hA, (2 * B_ * H_ * 2 + B_ * H_ * 4) / 4);
  for (int t = 0; t < S_; ++t)
    lstm_step<<<32, blk, 0, stream>>>(gx, Whh0b, (t & 1) ? hB : hA,
                                      (t & 1) ? hA : hB, cst, h0seq, t);

  // --- LSTM layer 1 ---
  wmma_gemm_nt<<<dim3(BS_ / 128, G4_ / 128, 1), blk, 0, stream>>>(
      h0seq, Wih1b, bc1, gx, nullptr, H_, H_, H_, G4_, 0, 0, 0, 1.0f);
  zero_u32<<<128, blk, 0, stream>>>((uint32_t*)hA, (2 * B_ * H_ * 2 + B_ * H_ * 4) / 4);
  for (int t = 0; t < S_; ++t)
    lstm_step<<<32, blk, 0, stream>>>(gx, Whh1b, (t & 1) ? hB : hA,
                                      (t & 1) ? hA : hB, cst, h1seq, t);

  // --- attention projections (bf16 outputs) ---
  wmma_gemm_nt<<<dim3(BS_ / 128, H_ / 128, 1), blk, 0, stream>>>(
      h1seq, Wqb, bq, nullptr, qb, H_, H_, H_, H_, 0, 0, 0, 1.0f);
  wmma_gemm_nt<<<dim3(BS_ / 128, H_ / 128, 1), blk, 0, stream>>>(
      h1seq, Wkb, bk, nullptr, kb, H_, H_, H_, H_, 0, 0, 0, 1.0f);
  wmma_gemm_nt<<<dim3(BS_ / 128, H_ / 128, 1), blk, 0, stream>>>(
      h1seq, Wvb, bv, nullptr, vb, H_, H_, H_, H_, 0, 0, 0, 1.0f);

  // v^T per batch for the attn@v GEMM
  transpose_bf16<<<dim3(H_ / 32, S_ / 32, B_), dim3(32, 8), 0, stream>>>(vb, vT);

  // scores = (q @ k^T) / sqrt(H) : batched, M=N=K=1024, batch=16
  wmma_gemm_nt<<<dim3(S_ / 128, S_ / 128, B_), blk, 0, stream>>>(
      qb, kb, nullptr, scores, nullptr, H_, H_, H_, S_,
      (long long)S_ * H_, (long long)S_ * H_, (long long)S_ * S_, 0.03125f);

  softmax_rows<<<BS_, blk, 0, stream>>>(scores, attn);

  // ctx = attn @ v : batched, A=attn [1024x1024], B=vT (n=h rows, k=s)
  wmma_gemm_nt<<<dim3(S_ / 128, H_ / 128, B_), blk, 0, stream>>>(
      attn, vT, nullptr, nullptr, ctx, S_, S_, S_, H_,
      (long long)S_ * S_, (long long)S_ * H_, (long long)S_ * H_, 1.0f);

  // out = ctx @ Wproj^T : M=16384, N=512, K=1024 (fp32 out)
  wmma_gemm_nt<<<dim3(BS_ / 128, D_ / 128, 1), blk, 0, stream>>>(
      ctx, Wprojb, nullptr, projo, nullptr, H_, H_, H_, D_, 0, 0, 0, 1.0f);

  // residual + LayerNorm -> d_out
  add_layernorm<<<BS_, blk, 0, stream>>>(projo, xf, gam, bet, outp);
}
